// TripletInteraction_82660940578989
// MI455X (gfx1250) — matrix operand order repaired and checked by reading, not compiled, not run on hardware
//
#include <hip/hip_runtime.h>
#include <math.h>

typedef __attribute__((ext_vector_type(16))) __bf16 v16bf;
typedef __attribute__((ext_vector_type(8)))  __bf16 v8bf;
typedef __attribute__((ext_vector_type(4)))  __bf16 v4bf;
typedef __attribute__((ext_vector_type(8)))  float  v8f;
typedef __attribute__((ext_vector_type(4)))  unsigned int u32x4;
typedef __attribute__((ext_vector_type(8)))  int i32x8;
typedef __attribute__((ext_vector_type(4)))  int i32x4;

#define E_NUM 65536
#define T_NUM 262144
#define INV_SQRT2F 0.70710678118654752440f

// ---- WMMA fragment layouts (CDNA5 ISA 7.12.2, wave32) ----
// A 16x32 bf16: lane<16 holds M=lane, K = {koff..koff+7, koff+16..koff+23}, koff = (lane>=16)*8
// B 32x16 bf16: lane holds N=lane&15, K = {koff..koff+7, koff+16..koff+23}, koff = (lane>=16)*8
// C/D f32 16x16: reg r, lane<16 -> (M=r, N=lane); lane>=16 -> (M=r+8, N=lane-16)

__device__ __forceinline__ v16bf cat8(v8bf lo, v8bf hi) {
    return __builtin_shufflevector(lo, hi, 0, 1, 2, 3, 4, 5, 6, 7, 8, 9, 10, 11, 12, 13, 14, 15);
}
// row = pointer to K-contiguous data for this lane's M row (stride-padded LDS)
__device__ __forceinline__ v16bf load_a_frag(const __bf16* row, int lane) {
    const int koff = (lane >> 4) << 3;
    v8bf lo = *(const v8bf*)(row + koff);
    v8bf hi = *(const v8bf*)(row + koff + 16);
    return cat8(lo, hi);
}
// rowN = pointer to K-contiguous data for this lane's N column
__device__ __forceinline__ v16bf load_b_frag(const __bf16* rowN, int lane) {
    const int koff = (lane >> 4) << 3;
    v8bf lo = *(const v8bf*)(rowN + koff);
    v8bf hi = *(const v8bf*)(rowN + koff + 16);
    return cat8(lo, hi);
}

__device__ __forceinline__ float silu_f(float x) { return x / (1.f + __expf(-x)); }

__device__ __forceinline__ v8f wmma_bf16(v16bf a, v16bf b, v8f c) {
    return __builtin_amdgcn_wmma_f32_16x16x32_bf16(false, a, false, b, (short)0, c, false, false);
}

__device__ __forceinline__ unsigned lds_off(const void* p) {
    // flat LDS pointers carry the LDS byte offset in their low 32 bits
    return (unsigned)(unsigned long long)p;
}

// ---- Tensor Data Mover: 2D bf16 tile load, global -> padded LDS (D# per ISA 8.3/8.4) ----
// tile_w elems contiguous (dim0), tile_h rows (dim1); pad_icode: 8<<code bytes per pad interval,
// pad_dw: DWORDs of LDS padding appended per interval (row stride = tile_w*2 + pad_dw*4 bytes).
__device__ __forceinline__ void tdm_load_2d_bf16(
    const __bf16* gsrc, unsigned ldsoff,
    unsigned tile_w, unsigned tile_h,
    unsigned tensor_w, unsigned tensor_h, unsigned row_stride,
    unsigned pad_icode, unsigned pad_dw)
{
    unsigned long long ga = (unsigned long long)gsrc;
    u32x4 g0;
    g0[0] = 1u;                                         // count=1, user descriptor
    g0[1] = ldsoff;                                     // lds_addr
    g0[2] = (unsigned)(ga & 0xffffffffu);               // global_addr[31:0]
    g0[3] = (unsigned)((ga >> 32) & 0x01ffffffu) | (2u << 30);  // addr[56:32] | type=2
    unsigned d0 = (1u << 16)                            // data_size=1 -> 2 bytes
                | (1u << 20)                            // pad_enable
                | (pad_icode << 22)
                | ((pad_dw - 1u) << 25);
    i32x8 g1;
    g1[0] = (int)d0;
    g1[1] = (int)((tensor_w & 0xffffu) << 16);          // atomic_addr=0 | tensor_dim0 lo
    g1[2] = (int)((tensor_w >> 16) | ((tensor_h & 0xffffu) << 16));
    g1[3] = (int)((tensor_h >> 16) | (tile_w << 16));   // tensor_dim1 hi | tile_dim0
    g1[4] = (int)(tile_h & 0xffffu);                    // tile_dim1 | tile_dim2=0
    g1[5] = (int)row_stride;                            // tensor_dim0_stride[31:0]
    g1[6] = 0;                                          // stride hi | dim1_stride lo
    g1[7] = 0;
    i32x4 z4 = {0, 0, 0, 0};
    i32x8 z8 = {0, 0, 0, 0, 0, 0, 0, 0};
    __builtin_amdgcn_tensor_load_to_lds(g0, g1, z4, z4, z8, 0);
}

// ---------------- utility kernels ----------------
__global__ void cvt_f32_bf16(const float* __restrict__ in, __bf16* __restrict__ out, int n) {
    int i = blockIdx.x * blockDim.x + threadIdx.x;
    if (i < n) out[i] = (__bf16)in[i];
}
// in: rows x cols (f32) -> out: cols x rows (bf16), i.e. [N][K] weight layout
__global__ void cvt_transpose_bf16(const float* __restrict__ in, __bf16* __restrict__ out,
                                   int rows, int cols) {
    int i = blockIdx.x * blockDim.x + threadIdx.x;
    if (i < rows * cols) {
        int r = i / cols, c = i % cols;
        out[(size_t)c * rows + r] = (__bf16)in[i];
    }
}
__global__ void zero_f32(float* __restrict__ p, int n) {
    int i = blockIdx.x * blockDim.x + threadIdx.x;
    if (i < n) p[i] = 0.f;
}

// ---------------- K1: m_kt = bf16( silu(m_st@W_mkt) * (rbf@W_rbf) * s_rbf ) ----------------
// 128 thr (4 waves), tile 64 rows x 64 cols, grid (512/64, E/64). Wmkt_t is [N=512][K=512].
__global__ __launch_bounds__(128) void k1_edge_mlp(
    const float* __restrict__ m_st, const float* __restrict__ rbf,
    const float* __restrict__ W_rbf, const __bf16* __restrict__ Wmkt_t,
    const float* __restrict__ s_rbf, __bf16* __restrict__ m_kt_bf)
{
    __shared__ __align__(16) __bf16 As[64 * 40];   // [row][k], stride 40
    __shared__ __align__(16) __bf16 Bt[64 * 40];   // [n][k],  stride 40 (TDM-padded)
    __shared__ __align__(16) float  Wr[16][64];
    __shared__ __align__(16) float  Rb[64][16];
    const int tid = threadIdx.x, lane = tid & 31, wave = tid >> 5;
    const int row0 = blockIdx.y * 64, col0 = blockIdx.x * 64;

    for (int i = tid; i < 16 * 64; i += 128)
        Wr[i >> 6][i & 63] = W_rbf[(size_t)(i >> 6) * 512 + col0 + (i & 63)];
    for (int i = tid; i < 64 * 16; i += 128)
        Rb[i >> 4][i & 15] = rbf[(size_t)(row0 + (i >> 4)) * 16 + (i & 15)];

    v8f acc[4] = {};
    for (int k0 = 0; k0 < 512; k0 += 32) {
        __syncthreads();
        if (wave == 0)   // 32x64 tile of Wmkt_t, rows n=col0.., k=k0.. ; pad 64B rows by 16B
            tdm_load_2d_bf16(Wmkt_t + (size_t)col0 * 512 + k0, lds_off(Bt),
                             32, 64, 512, 512, 512, 3, 4);
        for (int i = tid * 4; i < 64 * 32; i += 128 * 4) {
            int r = i >> 5, c = i & 31;
            const float4 v = *(const float4*)&m_st[(size_t)(row0 + r) * 512 + k0 + c];
            v4bf b; b[0] = (__bf16)v.x; b[1] = (__bf16)v.y; b[2] = (__bf16)v.z; b[3] = (__bf16)v.w;
            *(v4bf*)&As[r * 40 + c] = b;
        }
        if (k0 + 32 < 512)
            __builtin_prefetch(&m_st[(size_t)(row0 + (tid & 63)) * 512 + k0 + 32], 0, 1);
        __builtin_amdgcn_s_wait_tensorcnt(0);
        __syncthreads();
        const v16bf af = load_a_frag(&As[(wave * 16 + (lane & 15)) * 40], lane);
#pragma unroll
        for (int nt = 0; nt < 4; ++nt) {
            const v16bf bfr = load_b_frag(&Bt[(nt * 16 + (lane & 15)) * 40], lane);
            acc[nt] = wmma_bf16(af, bfr, acc[nt]);
        }
    }
    const float sr = *s_rbf;
    const int rbase = wave * 16 + ((lane >> 4) << 3);
    const int ncol = lane & 15;
#pragma unroll
    for (int nt = 0; nt < 4; ++nt) {
        float wc[16];
#pragma unroll
        for (int i = 0; i < 16; ++i) wc[i] = Wr[i][nt * 16 + ncol];
#pragma unroll
        for (int r = 0; r < 8; ++r) {
            int lrow = rbase + r;
            const float4* rbp = (const float4*)&Rb[lrow][0];
            float4 q0 = rbp[0], q1 = rbp[1], q2 = rbp[2], q3 = rbp[3];
            float rm = q0.x * wc[0] + q0.y * wc[1] + q0.z * wc[2] + q0.w * wc[3]
                     + q1.x * wc[4] + q1.y * wc[5] + q1.z * wc[6] + q1.w * wc[7]
                     + q2.x * wc[8] + q2.y * wc[9] + q2.z * wc[10] + q2.w * wc[11]
                     + q3.x * wc[12] + q3.y * wc[13] + q3.z * wc[14] + q3.w * wc[15];
            float h = silu_f(acc[nt][r]);
            m_kt_bf[(size_t)(row0 + lrow) * 512 + col0 + nt * 16 + ncol] = (__bf16)(h * rm * sr);
        }
    }
}

// ---------------- K2: m_down = bf16( silu(m_kt @ W_down) )  (E,64). Wdown_t is [64][512]. ----------------
__global__ __launch_bounds__(128) void k2_down(
    const __bf16* __restrict__ m_kt_bf, const __bf16* __restrict__ Wdown_t,
    __bf16* __restrict__ m_down_bf)
{
    __shared__ __align__(16) __bf16 As[64 * 40];
    __shared__ __align__(16) __bf16 Bt[64 * 40];
    const int tid = threadIdx.x, lane = tid & 31, wave = tid >> 5;
    const int row0 = blockIdx.x * 64;
    v8f acc[4] = {};
    for (int k0 = 0; k0 < 512; k0 += 32) {
        __syncthreads();
        if (wave == 0)
            tdm_load_2d_bf16(Wdown_t + k0, lds_off(Bt), 32, 64, 512, 64, 512, 3, 4);
        for (int i = tid * 8; i < 64 * 32; i += 128 * 8) {
            int r = i >> 5, c = i & 31;
            *(v8bf*)&As[r * 40 + c] = *(const v8bf*)&m_kt_bf[(size_t)(row0 + r) * 512 + k0 + c];
        }
        __builtin_amdgcn_s_wait_tensorcnt(0);
        __syncthreads();
        const v16bf af = load_a_frag(&As[(wave * 16 + (lane & 15)) * 40], lane);
#pragma unroll
        for (int nt = 0; nt < 4; ++nt) {
            const v16bf bfr = load_b_frag(&Bt[(nt * 16 + (lane & 15)) * 40], lane);
            acc[nt] = wmma_bf16(af, bfr, acc[nt]);
        }
    }
    const int rbase = wave * 16 + ((lane >> 4) << 3);
    const int ncol = lane & 15;
#pragma unroll
    for (int nt = 0; nt < 4; ++nt)
#pragma unroll
        for (int r = 0; r < 8; ++r)
            m_down_bf[(size_t)(row0 + rbase + r) * 64 + nt * 16 + ncol] = (__bf16)silu_f(acc[nt][r]);
}

// ---------------- K3: per-triplet bilinear + fused segment-sum -> x (E,64) ----------------
// A(16x1024) = cbf (x) m_t built on the fly (packed bf16 mul); B = Wbil_t[64][1024] via TDM.
__global__ __launch_bounds__(128) void k3_triplet(
    const float* __restrict__ cbf, const int* __restrict__ id3_kt,
    const int* __restrict__ id3_st, const __bf16* __restrict__ m_down_bf,
    const __bf16* __restrict__ Wbil_t, const float* __restrict__ s_cbf,
    float* __restrict__ x_out)
{
    __shared__ __align__(16) __bf16 Ms[4][16 * 64];
    __shared__ __align__(16) __bf16 Cb[4][16][16];
    __shared__ int St[4][16];
    __shared__ __align__(16) __bf16 Bt[64 * 40];
    const int tid = threadIdx.x, lane = tid & 31, wave = tid >> 5;
    const int t0 = (blockIdx.x * 4 + wave) * 16;

    for (int i = lane * 8; i < 16 * 64; i += 32 * 8) {
        int r = i >> 6, c = i & 63;
        *(v8bf*)&Ms[wave][r * 64 + c] =
            *(const v8bf*)&m_down_bf[(size_t)id3_kt[t0 + r] * 64 + c];
    }
    for (int i = lane; i < 16 * 16; i += 32)
        Cb[wave][i >> 4][i & 15] = (__bf16)cbf[(size_t)(t0 + (i >> 4)) * 16 + (i & 15)];
    if (lane < 16) St[wave][lane] = id3_st[t0 + lane];

    const int am = lane & 15;
    const int koff = (lane >> 4) << 3;
    v8f acc[4] = {};
    for (int k0 = 0; k0 < 1024; k0 += 32) {
        __syncthreads();
        if (wave == 0)
            tdm_load_2d_bf16(Wbil_t + k0, lds_off(Bt), 32, 64, 1024, 64, 1024, 3, 4);
        __builtin_amdgcn_s_wait_tensorcnt(0);
        __syncthreads();
        const int ka = k0 + koff, kb = ka + 16;          // both multiples of 8
        v8bf m0 = *(const v8bf*)&Ms[wave][am * 64 + (ka & 63)];
        v8bf m1 = *(const v8bf*)&Ms[wave][am * 64 + (kb & 63)];
        __bf16 c0 = Cb[wave][am][ka >> 6];
        __bf16 c1 = Cb[wave][am][kb >> 6];
        const v16bf af = cat8(m0 * c0, m1 * c1);         // packed bf16 scale
#pragma unroll
        for (int nt = 0; nt < 4; ++nt) {
            const v16bf bfr = load_b_frag(&Bt[(nt * 16 + (lane & 15)) * 40], lane);
            acc[nt] = wmma_bf16(af, bfr, acc[nt]);
        }
    }
    const float sc = *s_cbf;
    const int rbase = (lane >> 4) << 3;
    const int ncol = lane & 15;
#pragma unroll
    for (int nt = 0; nt < 4; ++nt)
#pragma unroll
        for (int r = 0; r < 8; ++r) {
            int edge = St[wave][rbase + r];
            __hip_atomic_fetch_add(&x_out[(size_t)edge * 64 + nt * 16 + ncol],
                                   acc[nt][r] * sc,
                                   __ATOMIC_RELAXED, __HIP_MEMORY_SCOPE_AGENT);
        }
}

// ---------------- K4: out = (silu(x@W_st) + silu(x[idx_swap]@W_ts)) * 1/sqrt(2) ----------------
// Wst_t/Wts_t are [512][64]; B tiles via TDM (pad 128B rows by 16B -> stride 72 elems).
__global__ __launch_bounds__(128) void k4_out(
    const float* __restrict__ x, const int* __restrict__ idx_swap,
    const __bf16* __restrict__ Wst_t, const __bf16* __restrict__ Wts_t,
    float* __restrict__ out)
{
    __shared__ __align__(16) __bf16 A1[64 * 72];
    __shared__ __align__(16) __bf16 A2[64 * 72];
    __shared__ __align__(16) __bf16 B1t[64 * 72];
    __shared__ __align__(16) __bf16 B2t[64 * 72];
    const int tid = threadIdx.x, lane = tid & 31, wave = tid >> 5;
    const int row0 = blockIdx.y * 64, col0 = blockIdx.x * 64;

    if (wave == 0) {
        tdm_load_2d_bf16(Wst_t + (size_t)col0 * 64, lds_off(B1t), 64, 64, 64, 512, 64, 4, 4);
        tdm_load_2d_bf16(Wts_t + (size_t)col0 * 64, lds_off(B2t), 64, 64, 64, 512, 64, 4, 4);
    }
    for (int i = tid * 4; i < 64 * 64; i += 128 * 4) {
        int r = i >> 6, c = i & 63;
        const float4 va = *(const float4*)&x[(size_t)(row0 + r) * 64 + c];
        v4bf a; a[0] = (__bf16)va.x; a[1] = (__bf16)va.y; a[2] = (__bf16)va.z; a[3] = (__bf16)va.w;
        *(v4bf*)&A1[r * 72 + c] = a;
        const float4 vb = *(const float4*)&x[(size_t)idx_swap[row0 + r] * 64 + c];
        v4bf b; b[0] = (__bf16)vb.x; b[1] = (__bf16)vb.y; b[2] = (__bf16)vb.z; b[3] = (__bf16)vb.w;
        *(v4bf*)&A2[r * 72 + c] = b;
    }
    __builtin_amdgcn_s_wait_tensorcnt(0);
    __syncthreads();

    v8f acc1[4] = {}, acc2[4] = {};
    const int am = wave * 16 + (lane & 15);
#pragma unroll
    for (int k0 = 0; k0 < 64; k0 += 32) {
        const v16bf a1f = load_a_frag(&A1[am * 72 + k0], lane);
        const v16bf a2f = load_a_frag(&A2[am * 72 + k0], lane);
#pragma unroll
        for (int nt = 0; nt < 4; ++nt) {
            const int bn = nt * 16 + (lane & 15);
            const v16bf b1f = load_b_frag(&B1t[bn * 72 + k0], lane);
            const v16bf b2f = load_b_frag(&B2t[bn * 72 + k0], lane);
            acc1[nt] = wmma_bf16(a1f, b1f, acc1[nt]);
            acc2[nt] = wmma_bf16(a2f, b2f, acc2[nt]);
        }
    }
    const int rbase = wave * 16 + ((lane >> 4) << 3);
    const int ncol = lane & 15;
#pragma unroll
    for (int nt = 0; nt < 4; ++nt)
#pragma unroll
        for (int r = 0; r < 8; ++r) {
            float v = (silu_f(acc1[nt][r]) + silu_f(acc2[nt][r])) * INV_SQRT2F;
            out[(size_t)(row0 + rbase + r) * 512 + col0 + nt * 16 + ncol] = v;
        }
}

// ---------------- host launch ----------------
extern "C" void kernel_launch(void* const* d_in, const int* in_sizes, int n_in,
                              void* d_out, int out_size, void* d_ws, size_t ws_size,
                              hipStream_t stream) {
    (void)in_sizes; (void)n_in; (void)out_size; (void)ws_size;
    const float* m_st   = (const float*)d_in[0];
    const float* rbf    = (const float*)d_in[1];
    const float* cbf    = (const float*)d_in[2];
    const int*   idx_sw = (const int*)d_in[3];
    const int*   id3_kt = (const int*)d_in[4];
    const int*   id3_st = (const int*)d_in[5];
    /* d_in[6] id3_ragged_idx: unused (atomics over sorted segments) */
    const float* W_mkt  = (const float*)d_in[7];
    const float* W_rbf  = (const float*)d_in[8];
    const float* W_down = (const float*)d_in[9];
    const float* W_bil  = (const float*)d_in[10];
    const float* W_st   = (const float*)d_in[11];
    const float* W_ts   = (const float*)d_in[12];
    const float* s_rbf  = (const float*)d_in[13];
    const float* s_cbf  = (const float*)d_in[14];

    char* ws = (char*)d_ws;
    size_t off = 0;
    __bf16* Wmkt_t  = (__bf16*)(ws + off); off += (size_t)512 * 512 * 2;
    __bf16* Wdown_t = (__bf16*)(ws + off); off += (size_t)64 * 512 * 2;
    __bf16* Wbil_t  = (__bf16*)(ws + off); off += (size_t)64 * 1024 * 2;
    __bf16* Wst_t   = (__bf16*)(ws + off); off += (size_t)512 * 64 * 2;
    __bf16* Wts_t   = (__bf16*)(ws + off); off += (size_t)512 * 64 * 2;
    __bf16* m_kt_bf = (__bf16*)(ws + off); off += (size_t)E_NUM * 512 * 2;
    __bf16* m_dn_bf = (__bf16*)(ws + off); off += (size_t)E_NUM * 64 * 2;
    float*  x       = (float*)(ws + off);  off += (size_t)E_NUM * 64 * 4;

    cvt_transpose_bf16<<<(512 * 512 + 255) / 256, 256, 0, stream>>>(W_mkt, Wmkt_t, 512, 512);
    cvt_transpose_bf16<<<(512 * 64 + 255) / 256, 256, 0, stream>>>(W_down, Wdown_t, 512, 64);
    cvt_transpose_bf16<<<(1024 * 64 + 255) / 256, 256, 0, stream>>>(W_bil, Wbil_t, 1024, 64);
    cvt_transpose_bf16<<<(64 * 512 + 255) / 256, 256, 0, stream>>>(W_st, Wst_t, 64, 512);
    cvt_transpose_bf16<<<(64 * 512 + 255) / 256, 256, 0, stream>>>(W_ts, Wts_t, 64, 512);
    zero_f32<<<(E_NUM * 64 + 255) / 256, 256, 0, stream>>>(x, E_NUM * 64);

    k1_edge_mlp<<<dim3(512 / 64, E_NUM / 64), 128, 0, stream>>>(m_st, rbf, W_rbf, Wmkt_t, s_rbf, m_kt_bf);
    k2_down<<<E_NUM / 64, 128, 0, stream>>>(m_kt_bf, Wdown_t, m_dn_bf);
    k3_triplet<<<T_NUM / 64, 128, 0, stream>>>(cbf, id3_kt, id3_st, m_dn_bf, Wbil_t, s_cbf, x);
    k4_out<<<dim3(512 / 64, E_NUM / 64), 128, 0, stream>>>(x, idx_sw, Wst_t, Wts_t, (float*)d_out);
}
